// OptimizedMultiHeadAttention_962072674699
// MI455X (gfx1250) — matrix operand order repaired and checked by reading, not compiled, not run on hardware
//
#include <hip/hip_runtime.h>
#include <hip/hip_bf16.h>

typedef __attribute__((ext_vector_type(16))) __bf16 v16bf;
typedef __attribute__((ext_vector_type(8)))  __bf16 v8bf;
typedef __attribute__((ext_vector_type(8)))  float  v8f;

typedef unsigned int u32x4 __attribute__((ext_vector_type(4)));
typedef int          i32x4 __attribute__((ext_vector_type(4)));
typedef int          i32x8 __attribute__((ext_vector_type(8)));

#define D_MODEL   1024
#define NUM_HEADS 16
#define D_K       64
#define SEQ       2048
#define BATCH     2
#define ROWS      (BATCH * SEQ)   // 4096 total token rows

static __device__ __forceinline__ v8f wmma_bf16(v16bf a, v16bf b, v8f c) {
  // D = A(16x32 bf16) x B(32x16 bf16) + C(16x16 f32)
  return __builtin_amdgcn_wmma_f32_16x16x32_bf16(false, a, false, b, (short)0, c,
                                                 false, false);
}

// ---------------------------------------------------------------------------
// f32 -> bf16 elementwise convert (grid-stride)
// ---------------------------------------------------------------------------
__global__ void convert_f32_bf16(const float* __restrict__ in,
                                 __bf16* __restrict__ out, int n) {
  int i = blockIdx.x * blockDim.x + threadIdx.x;
  int stride = gridDim.x * blockDim.x;
  for (; i < n; i += stride) out[i] = (__bf16)in[i];
}

// ---------------------------------------------------------------------------
// Build a 16x32 bf16 A-fragment (row = lane&15).
// grp=lane>>4; k = grp*16+j
// => two contiguous 8-element (16B) chunks at offsets grp*16 and grp*16+8.
// Works for both global and LDS source pointers.
// ---------------------------------------------------------------------------
static __device__ __forceinline__ v16bf load_a_frag(const __bf16* rowptr, int grp) {
  const __bf16* p = rowptr + grp * 16;
  v8bf a0 = *(const v8bf*)(p);
  v8bf a1 = *(const v8bf*)(p + 8);
  v16bf a;
#pragma unroll
  for (int i = 0; i < 8; ++i) { a[i] = a0[i]; a[8 + i] = a1[i]; }
  return a;
}

// ---------------------------------------------------------------------------
// Stage a 16 x 1024 bf16 tile (32 KB) of row-major A into LDS offset 0.
// Preferred path: one TDM tensor_load_to_lds issued by wave 0, then
// s_wait_tensorcnt 0. Fallback: cooperative vector copy.
// The trailing "memory" clobber is REQUIRED: the TDM builtin writes LDS
// behind the compiler's back (lds_addr passed as an integer), so without it
// the compiler treats the staged tile as undef and deletes the ds_loads.
// Caller must __syncthreads() after this returns.
// ---------------------------------------------------------------------------
static __device__ __forceinline__ void stage_a_tile(__bf16* lds_tile,
                                                    const __bf16* gsrc) {
#if __has_builtin(__builtin_amdgcn_tensor_load_to_lds) && \
    __has_builtin(__builtin_amdgcn_s_wait_tensorcnt)
  if (threadIdx.x < 32) {   // wave 0 issues the TDM descriptor
    unsigned long long ga = (unsigned long long)(const void*)gsrc;
    u32x4 g0;
    g0[0] = 1u;                                      // count=1, user mode
    g0[1] = 0u;                                      // lds_addr = 0 (tile is 1st shared obj)
    g0[2] = (unsigned)(ga & 0xFFFFFFFFu);            // global_addr[31:0]
    g0[3] = (unsigned)((ga >> 32) & 0x1FFFFFFu)      // global_addr[56:32]
          | (2u << 30);                              // type = 2 ("image")
    i32x8 g1;
    g1[0] = (1 << 16);                               // data_size = 1 -> 2 bytes
    g1[1] = (int)((D_MODEL & 0xFFFF) << 16);         // tensor_dim0 = 1024
    g1[2] = (int)(16u << 16);                        // tensor_dim1 = 16 rows
    g1[3] = (int)((unsigned)D_MODEL << 16);          // tile_dim0 = 1024
    g1[4] = 16;                                      // tile_dim1 = 16, tile_dim2 = 0
    g1[5] = D_MODEL;                                 // tensor_dim0_stride = 1024
    g1[6] = 0;
    g1[7] = 0;
    i32x4 gz = {0, 0, 0, 0};
#if __clang_major__ >= 23
    i32x8 gz8 = {0, 0, 0, 0, 0, 0, 0, 0};
    __builtin_amdgcn_tensor_load_to_lds(g0, g1, gz, gz, gz8, 0);
#else
    __builtin_amdgcn_tensor_load_to_lds(g0, g1, gz, gz, 0);
#endif
    __builtin_amdgcn_s_wait_tensorcnt(0);
  }
  // Tell the compiler LDS has been written (TDM is invisible to alias analysis).
  asm volatile("" : : "s"(lds_tile) : "memory");
#else
  // cooperative fallback: 256 threads x 16B vector copies
  const v8bf* s = (const v8bf*)gsrc;
  v8bf* d = (v8bf*)lds_tile;
  for (int i = threadIdx.x; i < 16 * D_MODEL / 8; i += blockDim.x) d[i] = s[i];
#endif
}

// ---------------------------------------------------------------------------
// QKV projection: Q/K/V[s][o] = sum_i x[s][i] * W[o][i]  (+ bias at writeout)
// Block = 256 threads (8 waves). All 8 waves share one 16x1024 A tile staged
// in LDS via TDM; wave w computes head h = blockIdx.y*8 + w (16x64 out tile).
// grid = (256 row-tiles, 2 head-halves, 3 matrices)
// Q,K stored [B,H,S,Dk] bf16 ; V stored transposed [B,H,Dk,S] bf16.
// ---------------------------------------------------------------------------
__global__ void proj_qkv(const __bf16* __restrict__ xb,
                         const __bf16* __restrict__ wqb,
                         const __bf16* __restrict__ wkb,
                         const __bf16* __restrict__ wvb,
                         const float* __restrict__ bq,
                         const float* __restrict__ bk,
                         const float* __restrict__ bv,
                         __bf16* __restrict__ Qb,
                         __bf16* __restrict__ Kb,
                         __bf16* __restrict__ Vtb) {
  __shared__ __align__(32) __bf16 atile[16 * D_MODEL];   // 32 KB, LDS offset 0

  const int lane = threadIdx.x & 31;
  const int grp  = lane >> 4;
  const int mn   = lane & 15;
  const int w    = threadIdx.x >> 5;     // wave id 0..7
  const int mt   = blockIdx.x;           // row tile 0..255
  const int h    = blockIdx.y * 8 + w;   // head == 64-wide out tile
  const int mat  = blockIdx.z;           // 0=Q 1=K 2=V

  const __bf16* W    = (mat == 0) ? wqb : (mat == 1) ? wkb : wvb;
  const float*  bias = (mat == 0) ? bq  : (mat == 1) ? bk  : bv;

  stage_a_tile(atile, xb + (size_t)mt * 16 * D_MODEL);
  __syncthreads();

  const __bf16* Arow = atile + (size_t)mn * D_MODEL;   // LDS reads

  v8f acc[4] = {v8f{}, v8f{}, v8f{}, v8f{}};
  for (int k0 = 0; k0 < D_MODEL; k0 += 32) {
    v16bf a = load_a_frag(Arow + k0, grp);
#pragma unroll
    for (int t = 0; t < 4; ++t) {
      // B[k][n] = W[h*64 + t*16 + n][k0 + k]; lane holds n=mn, k = grp*16 + j
      const __bf16* bp = W + (size_t)(h * 64 + t * 16 + mn) * D_MODEL + k0 + grp * 16;
      v16bf bf = *(const v16bf*)bp;
      acc[t] = wmma_bf16(a, bf, acc[t]);
    }
  }

#pragma unroll
  for (int t = 0; t < 4; ++t) {
#pragma unroll
    for (int r = 0; r < 8; ++r) {
      int m = r + grp * 8;
      int s = mt * 16 + m;          // global token row
      int b = s >> 11, si = s & (SEQ - 1);
      int d = t * 16 + mn;
      float v = acc[t][r] + bias[h * 64 + d];
      __bf16 bv16 = (__bf16)v;
      if (mat == 2)
        Vtb[(((size_t)b * NUM_HEADS + h) * D_K + d) * SEQ + si] = bv16;
      else if (mat == 0)
        Qb[(((size_t)b * NUM_HEADS + h) * SEQ + si) * D_K + d] = bv16;
      else
        Kb[(((size_t)b * NUM_HEADS + h) * SEQ + si) * D_K + d] = bv16;
    }
  }
}

// ---------------------------------------------------------------------------
// Flash attention: one wave = one 16-query tile of one (b,h).
// Online softmax over 32-key tiles; scores & PV via bf16 WMMA, f32 state.
// grid = (128 q-tiles, 32 bh), block = 32.
// Writes context to attn[b][s][h*64+d] bf16 ([B,S,D] layout).
// ---------------------------------------------------------------------------
__global__ void flash_attn(const __bf16* __restrict__ Qb,
                           const __bf16* __restrict__ Kb,
                           const __bf16* __restrict__ Vtb,
                           __bf16* __restrict__ attnb) {
  __shared__ float plds[16 * 32];   // P tile staging (C-layout f32 -> A-layout bf16)
  const int lane = threadIdx.x & 31;
  const int grp  = lane >> 4;
  const int mn   = lane & 15;
  const int qt   = blockIdx.x;      // 0..127
  const int bh   = blockIdx.y;      // 0..31
  const int b = bh >> 4, h = bh & 15;

  const __bf16* Q  = Qb  + ((size_t)bh * SEQ + qt * 16) * D_K;
  const __bf16* K  = Kb  + (size_t)bh * SEQ * D_K;
  const __bf16* Vt = Vtb + (size_t)bh * D_K * SEQ;   // [Dk][S]

  // Q fragments for k=0..31 and k=32..63 (resident whole loop)
  v16bf qa[2];
#pragma unroll
  for (int kk = 0; kk < 2; ++kk)
    qa[kk] = load_a_frag(Q + (size_t)mn * D_K + kk * 32, grp);

  v8f o[4] = {v8f{}, v8f{}, v8f{}, v8f{}};   // 16x64 f32 accumulator
  float mi[8], li[8];
#pragma unroll
  for (int r = 0; r < 8; ++r) { mi[r] = -1e30f; li[r] = 0.0f; }

  const float scale = 0.125f;   // 1/sqrt(64)

  for (int kv = 0; kv < SEQ; kv += 32) {
    // prefetch next key/value tiles into cache while we do VALU softmax work
    if (kv + 32 < SEQ) {
      __builtin_prefetch(K + (size_t)(kv + 32 + mn) * D_K, 0, 3);
      __builtin_prefetch(Vt + (size_t)mn * SEQ + kv + 32, 0, 3);
    }

    // ---- scores: S[m][n] = Q[m][:] . K[kv+n][:] , two 16-col halves
    v8f s0 = v8f{}, s1 = v8f{};
#pragma unroll
    for (int kk = 0; kk < 2; ++kk) {
      const __bf16* b0p = K + (size_t)(kv + mn)      * D_K + kk * 32 + grp * 16;
      const __bf16* b1p = K + (size_t)(kv + 16 + mn) * D_K + kk * 32 + grp * 16;
      v16bf b0 = *(const v16bf*)b0p;
      v16bf b1 = *(const v16bf*)b1p;
      s0 = wmma_bf16(qa[kk], b0, s0);
      s1 = wmma_bf16(qa[kk], b1, s1);
    }

    // ---- online softmax per row (row m=r+8*grp lives in a 16-lane half)
#pragma unroll
    for (int r = 0; r < 8; ++r) {
      float x0 = s0[r] * scale, x1 = s1[r] * scale;
      float t = fmaxf(x0, x1);
#pragma unroll
      for (int mask = 1; mask < 16; mask <<= 1)
        t = fmaxf(t, __shfl_xor(t, mask, 32));
      float newm  = fmaxf(mi[r], t);
      float alpha = __expf(mi[r] - newm);
      mi[r] = newm;
      float p0 = __expf(x0 - newm);
      float p1 = __expf(x1 - newm);
      float rs = p0 + p1;
#pragma unroll
      for (int mask = 1; mask < 16; mask <<= 1)
        rs += __shfl_xor(rs, mask, 32);
      li[r] = li[r] * alpha + rs;
#pragma unroll
      for (int t4 = 0; t4 < 4; ++t4) o[t4][r] *= alpha;
      int m = r + grp * 8;
      plds[m * 32 + mn]      = p0;
      plds[m * 32 + 16 + mn] = p1;
    }
    __syncthreads();

    // ---- re-fragment P (16x32) into A layout as bf16
    v16bf pa;
#pragma unroll
    for (int i = 0; i < 16; ++i) {
      int k = grp * 16 + i;
      pa[i] = (__bf16)plds[mn * 32 + k];
    }
    __syncthreads();

    // ---- O += P x V ; B[k][n] = V[kv+k][t*16+n] = Vt[(t*16+n)*SEQ + kv + k]
#pragma unroll
    for (int t4 = 0; t4 < 4; ++t4) {
      const __bf16* vp = Vt + (size_t)(t4 * 16 + mn) * SEQ + kv + grp * 16;
      v16bf bf = *(const v16bf*)vp;
      o[t4] = wmma_bf16(pa, bf, o[t4]);
    }
  }

  // ---- normalize, write context [B,S,D] bf16
#pragma unroll
  for (int r = 0; r < 8; ++r) {
    float inv = 1.0f / li[r];
    int m = r + grp * 8;
    size_t row = (size_t)b * SEQ + qt * 16 + m;
#pragma unroll
    for (int t4 = 0; t4 < 4; ++t4)
      attnb[row * D_MODEL + h * 64 + t4 * 16 + mn] = (__bf16)(o[t4][r] * inv);
  }
}

// ---------------------------------------------------------------------------
// Output projection: out[s][o] = sum_i attn[s][i] * wo[o][i] + bo[o] (f32 out)
// Block = 256 threads (8 waves) sharing one TDM-staged 16x1024 A tile.
// grid = (256 row-tiles, 2 out-tile halves).
// ---------------------------------------------------------------------------
__global__ void proj_out(const __bf16* __restrict__ attnb,
                         const __bf16* __restrict__ wob,
                         const float* __restrict__ bo,
                         float* __restrict__ out) {
  __shared__ __align__(32) __bf16 atile[16 * D_MODEL];   // 32 KB, LDS offset 0

  const int lane = threadIdx.x & 31;
  const int grp  = lane >> 4;
  const int mn   = lane & 15;
  const int w    = threadIdx.x >> 5;
  const int mt   = blockIdx.x;
  const int nt   = blockIdx.y * 8 + w;   // 64-wide out tile

  stage_a_tile(atile, attnb + (size_t)mt * 16 * D_MODEL);
  __syncthreads();

  const __bf16* Arow = atile + (size_t)mn * D_MODEL;

  v8f acc[4] = {v8f{}, v8f{}, v8f{}, v8f{}};
  for (int k0 = 0; k0 < D_MODEL; k0 += 32) {
    v16bf a = load_a_frag(Arow + k0, grp);
#pragma unroll
    for (int t = 0; t < 4; ++t) {
      const __bf16* bp = wob + (size_t)(nt * 64 + t * 16 + mn) * D_MODEL + k0 + grp * 16;
      v16bf bf = *(const v16bf*)bp;
      acc[t] = wmma_bf16(a, bf, acc[t]);
    }
  }

#pragma unroll
  for (int t = 0; t < 4; ++t) {
#pragma unroll
    for (int r = 0; r < 8; ++r) {
      int m = r + grp * 8;
      int oc = nt * 64 + t * 16 + mn;
      out[(size_t)(mt * 16 + m) * D_MODEL + oc] = acc[t][r] + bo[oc];
    }
  }
}

// ---------------------------------------------------------------------------
// Host launcher
// ---------------------------------------------------------------------------
extern "C" void kernel_launch(void* const* d_in, const int* in_sizes, int n_in,
                              void* d_out, int out_size, void* d_ws, size_t ws_size,
                              hipStream_t stream) {
  const float* x  = (const float*)d_in[0];
  const float* wq = (const float*)d_in[1];
  const float* bq = (const float*)d_in[2];
  const float* wk = (const float*)d_in[3];
  const float* bk = (const float*)d_in[4];
  const float* wv = (const float*)d_in[5];
  const float* bv = (const float*)d_in[6];
  const float* wo = (const float*)d_in[7];
  const float* bo = (const float*)d_in[8];
  float* out = (float*)d_out;

  const size_t MB = (size_t)1 << 20;
  char* ws = (char*)d_ws;
  __bf16* xb   = (__bf16*)(ws + 0 * MB);   // 8 MB  [4096,1024]
  __bf16* wqb  = (__bf16*)(ws + 8 * MB);   // 2 MB
  __bf16* wkb  = (__bf16*)(ws + 10 * MB);  // 2 MB
  __bf16* wvb  = (__bf16*)(ws + 12 * MB);  // 2 MB
  __bf16* wob  = (__bf16*)(ws + 14 * MB);  // 2 MB
  __bf16* Qb   = (__bf16*)(ws + 16 * MB);  // 8 MB  [B,H,S,Dk]
  __bf16* Kb   = (__bf16*)(ws + 24 * MB);  // 8 MB  [B,H,S,Dk]
  __bf16* Vtb  = (__bf16*)(ws + 32 * MB);  // 8 MB  [B,H,Dk,S]
  __bf16* ctx  = (__bf16*)(ws + 40 * MB);  // 8 MB  [B,S,D]

  // 1) downconvert activations + weights to bf16
  convert_f32_bf16<<<dim3(1024), dim3(256), 0, stream>>>(x,  xb,  ROWS * D_MODEL);
  convert_f32_bf16<<<dim3(512),  dim3(256), 0, stream>>>(wq, wqb, D_MODEL * D_MODEL);
  convert_f32_bf16<<<dim3(512),  dim3(256), 0, stream>>>(wk, wkb, D_MODEL * D_MODEL);
  convert_f32_bf16<<<dim3(512),  dim3(256), 0, stream>>>(wv, wvb, D_MODEL * D_MODEL);
  convert_f32_bf16<<<dim3(512),  dim3(256), 0, stream>>>(wo, wob, D_MODEL * D_MODEL);

  // 2) fused QKV projection (TDM-staged A tile + WMMA GEMM)
  proj_qkv<<<dim3(ROWS / 16, 2, 3), dim3(256), 0, stream>>>(
      xb, wqb, wkb, wvb, bq, bk, bv, Qb, Kb, Vtb);

  // 3) flash attention (WMMA scores + WMMA PV, online softmax)
  flash_attn<<<dim3(SEQ / 16, BATCH * NUM_HEADS), dim3(32), 0, stream>>>(
      Qb, Kb, Vtb, ctx);

  // 4) output projection (TDM-staged A tile + WMMA GEMM, f32 output + bias)
  proj_out<<<dim3(ROWS / 16, 2), dim3(256), 0, stream>>>(ctx, wob, bo, out);
}
